// Model_5463198400658
// MI455X (gfx1250) — matrix operand order repaired and checked by reading, not compiled, not run on hardware
//
#include <hip/hip_runtime.h>
#include <hip/hip_bf16.h>
#include <stdint.h>

#define NN 50000
#define TT 8
#define DD 256
#define HH 128
#define EE 800000

#define APB 8            // nodes per attention block
#define AROWS (APB*TT)   // 64 rows (node,token)
#define LDK 264          // padded LDS row stride (ushorts) -> breaks bank collisions
#define LDC 132          // padded f32 C-tile stride (same 33792-byte footprint)
#define PNODES 64        // nodes per projection block

typedef __attribute__((ext_vector_type(16))) __bf16        v16bf;
typedef __attribute__((ext_vector_type(8)))  float         v8f;
typedef __attribute__((ext_vector_type(4)))  unsigned int  uint4v;
typedef __attribute__((ext_vector_type(2)))  unsigned int  uint2v;

union BF16Frag { v16bf v; uint4v q[2]; };

__device__ inline unsigned short f32_to_bf16(float f) {
    union { float f; uint32_t u; } x; x.f = f;
    uint32_t u = x.u;
    uint32_t r = u + 0x7FFFu + ((u >> 16) & 1u);   // round-to-nearest-even
    if ((u & 0x7F800000u) == 0x7F800000u) r = u;   // keep NaN/Inf (cndmask, branchless)
    return (unsigned short)(r >> 16);
}

__device__ inline uint2v pack_bf16x4(float4 f) {
    uint2v p;
    p.x = (uint32_t)f32_to_bf16(f.x) | ((uint32_t)f32_to_bf16(f.y) << 16);
    p.y = (uint32_t)f32_to_bf16(f.z) | ((uint32_t)f32_to_bf16(f.w) << 16);
    return p;
}

// Branch-free tanh: one v_exp_f32, no divergent exec in the hot loop.
__device__ inline float fast_tanh(float x) {
    float xc = fminf(fmaxf(x, -15.f), 15.f);
    float e = __expf(-2.f * xc);
    return __fdividef(1.f - e, 1.f + e);
}

// Transpose + convert 4 weight matrices [256x128] f32 -> [128x256] bf16 (K-contiguous for B frags)
__global__ __launch_bounds__(256)
void prep_weights(const float* __restrict__ w0, const float* __restrict__ w1,
                  const float* __restrict__ w2, const float* __restrict__ w3,
                  unsigned short* __restrict__ o0, unsigned short* __restrict__ o1,
                  unsigned short* __restrict__ o2, unsigned short* __restrict__ o3) {
    int idx = blockIdx.x * 256 + threadIdx.x;     // 4 * 32768 total
    int m = idx >> 15;
    int j = idx & 32767;
    int h = j >> 8;        // 0..127
    int d = j & 255;       // 0..255
    const float* w = (m == 0) ? w0 : (m == 1) ? w1 : (m == 2) ? w2 : w3;
    unsigned short* o = (m == 0) ? o0 : (m == 1) ? o1 : (m == 2) ? o2 : o3;
    o[h * DD + d] = f32_to_bf16(w[d * HH + h]);
}

// Fused additive-attention pooling for BOTH heads: x read from HBM once.
// Block: 256 threads = 8 waves; wave w -> head w/4, M-tile w%4 (16 rows, 64 rows = 8 nodes).
__global__ __launch_bounds__(256)
void attn_pool(const float* __restrict__ x,
               const unsigned short* __restrict__ WtC, const unsigned short* __restrict__ WtO,
               const float* __restrict__ bC, const float* __restrict__ vC,
               const float* __restrict__ bO, const float* __restrict__ vO,
               float* __restrict__ cpool, float* __restrict__ opool) {
    __shared__ unsigned short tA[AROWS * LDK];   // 33 KB bf16 x-tile
    __shared__ float e_s[2][AROWS];
    __shared__ float a_s[2][AROWS];

    const int tid = threadIdx.x;
    const long n0 = (long)blockIdx.x * APB;
    const long rowbase = n0 * TT;
    const long totrows = (long)NN * TT;
    const float4* __restrict__ x4 = (const float4*)x;

    // Stage x tile: coalesced float4 loads -> packed bf16x4 ds_store_b64
    for (int j = tid; j < AROWS * 64; j += 256) {
        const int row = j >> 6, c4 = j & 63;
        const long grow = rowbase + row;
        float4 f = make_float4(0.f, 0.f, 0.f, 0.f);
        if (grow < totrows) f = x4[grow * 64 + c4];
        *(uint2v*)&tA[row * LDK + c4 * 4] = pack_bf16x4(f);
    }
    __syncthreads();

    const int wave  = tid >> 5;
    const int lane  = tid & 31;
    const int head  = wave >> 2;
    const int mtile = wave & 3;
    const int lhalf = lane >> 4;
    const int lmod  = lane & 15;

    const unsigned short* __restrict__ Wt = head ? WtO : WtC;
    const float* __restrict__ bb = head ? bO : bC;
    const float* __restrict__ vv = head ? vO : vC;

    const int arow = mtile * 16 + lmod;

    // Hoist all 8 A-fragments (K=256) into VGPRs once; reused across the 8 N-tiles.
    BF16Frag afr[8];
#pragma unroll
    for (int ks = 0; ks < 8; ++ks) {
        const unsigned short* ap = &tA[arow * LDK + ks * 32 + lhalf * 8];
        afr[ks].q[0] = *(const uint4v*)(ap);
        afr[ks].q[1] = *(const uint4v*)(ap + 16);
    }

    float ep[8];
#pragma unroll
    for (int r = 0; r < 8; ++r) ep[r] = 0.f;

    for (int ntile = 0; ntile < 8; ++ntile) {
        v8f acc = {};
#pragma unroll
        for (int ks = 0; ks < 8; ++ks) {
            const int kk = ks * 32;
            BF16Frag b;
            const unsigned short* bp = &Wt[(ntile * 16 + lmod) * DD + kk + lhalf * 8];
            b.q[0] = *(const uint4v*)(bp);
            b.q[1] = *(const uint4v*)(bp + 16);
            acc = __builtin_amdgcn_wmma_f32_16x16x32_bf16(false, afr[ks].v, false, b.v,
                                                          (short)0, acc, false, false);
        }
        const int col = ntile * 16 + lmod;
        const float bvv = bb[col], vvv = vv[col];
#pragma unroll
        for (int r = 0; r < 8; ++r)
            ep[r] += fast_tanh(acc[r] + bvv) * vvv;   // branch-free, stays convergent
    }

    // Reduce over the 16 columns (lanes within half-wave); rows: M = mtile*16 + 8*lhalf + r
#pragma unroll
    for (int r = 0; r < 8; ++r) {
        float s = ep[r];
        s += __shfl_xor(s, 1, 32);
        s += __shfl_xor(s, 2, 32);
        s += __shfl_xor(s, 4, 32);
        s += __shfl_xor(s, 8, 32);
        ep[r] = s;
    }
    if (lmod == 0) {
        const int rb = mtile * 16 + lhalf * 8;
#pragma unroll
        for (int r = 0; r < 8; ++r) e_s[head][rb + r] = ep[r];
    }
    __syncthreads();

    // Softmax over T=8 per (head, node)
    if (tid < 16) {
        const int h = tid >> 3, node = tid & 7;
        float mx = -1e30f;
        for (int t = 0; t < TT; ++t) mx = fmaxf(mx, e_s[h][node * 8 + t]);
        float ex[TT], sum = 0.f;
        for (int t = 0; t < TT; ++t) { ex[t] = __expf(e_s[h][node * 8 + t] - mx); sum += ex[t]; }
        const float inv = 1.f / sum;
        for (int t = 0; t < TT; ++t) a_s[h][node * 8 + t] = ex[t] * inv;
    }
    __syncthreads();

    // Pooled sums in fp32, both heads in one pass, float4 in/out (x re-read hits L2/WGP$)
    const int g  = tid >> 6;      // 0..3: node subgroup
    const int c4 = tid & 63;      // float4 column
    for (int nb = 0; nb < APB; nb += 4) {
        const int node = nb + g;
        const long n = n0 + node;
        if (n >= NN) continue;
        float sx = 0.f, sy = 0.f, sz = 0.f, sw = 0.f;
        float ox = 0.f, oy = 0.f, oz = 0.f, ow = 0.f;
#pragma unroll
        for (int t = 0; t < TT; ++t) {
            const float4 xv = x4[(n * TT + t) * 64 + c4];
            const float ac = a_s[0][node * 8 + t];
            const float ao = a_s[1][node * 8 + t];
            sx += ac * xv.x; sy += ac * xv.y; sz += ac * xv.z; sw += ac * xv.w;
            ox += ao * xv.x; oy += ao * xv.y; oz += ao * xv.z; ow += ao * xv.w;
        }
        ((float4*)cpool)[n * 64 + c4] = make_float4(sx, sy, sz, sw);
        ((float4*)opool)[n * 64 + c4] = make_float4(ox, oy, oz, ow);
    }
}

// LGConv: one wave per edge; float4 gather from src row (L2-resident), atomic scatter to dst.
__global__ __launch_bounds__(256)
void lgconv(const float* __restrict__ xin, const int* __restrict__ eidx,
            const float* __restrict__ ew, float* __restrict__ yout) {
    const int e = blockIdx.x * 8 + (threadIdx.x >> 5);
    if (e >= EE) return;
    const int lane = threadIdx.x & 31;
    const int src = eidx[e];
    const int dst = eidx[EE + e];
    const float w = ew[e];
    const float4* xs = (const float4*)(xin + (long)src * DD);
    float* yd = yout + (long)dst * DD;
#pragma unroll
    for (int i = 0; i < 2; ++i) {
        const int c4 = lane + 32 * i;
        const float4 v = xs[c4];
        atomicAdd(&yd[c4 * 4 + 0], v.x * w);
        atomicAdd(&yd[c4 * 4 + 1], v.y * w);
        atomicAdd(&yd[c4 * 4 + 2], v.z * w);
        atomicAdd(&yd[c4 * 4 + 3], v.w * w);
    }
}

// out = relu?( (a+b) @ Wt^T + bias ) ; WMMA bf16, fused residual, LDS-staged coalesced stores.
__global__ __launch_bounds__(128)
void project(const float* __restrict__ a, const float* __restrict__ b,
             const unsigned short* __restrict__ Wt, const float* __restrict__ bias,
             float* __restrict__ out, int relu) {
    __shared__ unsigned short tmem[PNODES * LDK];     // 33792 B: bf16 A-tile, then f32 C-tile
    unsigned short* tA = tmem;
    float* tC = (float*)tmem;                          // [PNODES][LDC]

    const int tid = threadIdx.x;
    const long n0 = (long)blockIdx.x * PNODES;
    const float4* __restrict__ a4 = (const float4*)a;
    const float4* __restrict__ b4 = (const float4*)b;

    // Stage (a+b) tile as bf16 (float4 loads, packed b64 stores)
    for (int j = tid; j < PNODES * 64; j += 128) {
        const int row = j >> 6, c4 = j & 63;
        const long n = n0 + row;
        float4 f = make_float4(0.f, 0.f, 0.f, 0.f);
        if (n < NN) {
            const float4 fa = a4[n * 64 + c4];
            const float4 fb = b4[n * 64 + c4];
            f = make_float4(fa.x + fb.x, fa.y + fb.y, fa.z + fb.z, fa.w + fb.w);
        }
        *(uint2v*)&tA[row * LDK + c4 * 4] = pack_bf16x4(f);
    }
    __syncthreads();

    const int wave = tid >> 5, lane = tid & 31;
    const int lhalf = lane >> 4, lmod = lane & 15;
    const int arow = wave * 16 + lmod;

    // Hoist all A-fragments, then the LDS tile is dead -> reuse it for the f32 C-tile.
    BF16Frag afr[8];
#pragma unroll
    for (int ks = 0; ks < 8; ++ks) {
        const unsigned short* ap = &tA[arow * LDK + ks * 32 + lhalf * 8];
        afr[ks].q[0] = *(const uint4v*)(ap);
        afr[ks].q[1] = *(const uint4v*)(ap + 16);
    }
    __syncthreads();   // everyone finished reading A before C-tile overwrite

    for (int ntile = 0; ntile < 8; ++ntile) {
        v8f acc = {};
#pragma unroll
        for (int ks = 0; ks < 8; ++ks) {
            const int kk = ks * 32;
            BF16Frag bf;
            const unsigned short* bp = &Wt[(ntile * 16 + lmod) * DD + kk + lhalf * 8];
            bf.q[0] = *(const uint4v*)(bp);
            bf.q[1] = *(const uint4v*)(bp + 16);
            acc = __builtin_amdgcn_wmma_f32_16x16x32_bf16(false, afr[ks].v, false, bf.v,
                                                          (short)0, acc, false, false);
        }
        const int col = ntile * 16 + lmod;
        const float bs = bias[col];
#pragma unroll
        for (int r = 0; r < 8; ++r) {
            float vvv = acc[r] + bs;
            if (relu) vvv = fmaxf(vvv, 0.f);
            tC[(wave * 16 + lhalf * 8 + r) * LDC + col] = vvv;
        }
    }
    __syncthreads();

    // Coalesced float4 stores of the C-tile
    for (int j = tid; j < PNODES * 32; j += 128) {    // 32 float4-cols per row
        const int row = j >> 5, c4 = j & 31;
        const long n = n0 + row;
        if (n < NN)
            *(float4*)&out[n * HH + c4 * 4] = *(float4*)&tC[row * LDC + c4 * 4];
    }
}

extern "C" void kernel_launch(void* const* d_in, const int* in_sizes, int n_in,
                              void* d_out, int out_size, void* d_ws, size_t ws_size,
                              hipStream_t stream) {
    const float* x     = (const float*)d_in[0];
    const int*   eidx  = (const int*)  d_in[1];
    const float* ew    = (const float*)d_in[2];
    const float* attcW = (const float*)d_in[3];
    const float* attcb = (const float*)d_in[4];
    const float* attcv = (const float*)d_in[5];
    const float* attoW = (const float*)d_in[6];
    const float* attob = (const float*)d_in[7];
    const float* attov = (const float*)d_in[8];
    const float* Wc    = (const float*)d_in[9];
    const float* bc    = (const float*)d_in[10];
    const float* Wo    = (const float*)d_in[11];
    const float* bo    = (const float*)d_in[12];
    float* outp = (float*)d_out;

    char* ws = (char*)d_ws;
    unsigned short* WtC  = (unsigned short*)ws; ws += 65536;
    unsigned short* WtO  = (unsigned short*)ws; ws += 65536;
    unsigned short* WtPc = (unsigned short*)ws; ws += 65536;
    unsigned short* WtPo = (unsigned short*)ws; ws += 65536;
    const size_t fsz = (size_t)NN * DD * sizeof(float);
    float* cpool = (float*)ws; ws += fsz;
    float* opool = (float*)ws; ws += fsz;
    float* t1    = (float*)ws; ws += fsz;
    float* t2    = (float*)ws; ws += fsz;

    prep_weights<<<512, 256, 0, stream>>>(attcW, attoW, Wc, Wo, WtC, WtO, WtPc, WtPo);

    attn_pool<<<(NN + APB - 1) / APB, 256, 0, stream>>>(
        x, WtC, WtO, attcb, attcv, attob, attov, cpool, opool);

    // head c: two LGConv hops + residual, projected to center
    hipMemsetAsync(t1, 0, fsz, stream);
    lgconv<<<(EE + 7) / 8, 256, 0, stream>>>(cpool, eidx, ew, t1);
    hipMemsetAsync(t2, 0, fsz, stream);
    lgconv<<<(EE + 7) / 8, 256, 0, stream>>>(t1, eidx, ew, t2);
    project<<<(NN + PNODES - 1) / PNODES, 128, 0, stream>>>(t2, cpool, WtPc, bc, outp, 0);

    // head o: same, with ReLU, into second half of d_out
    hipMemsetAsync(t1, 0, fsz, stream);
    lgconv<<<(EE + 7) / 8, 256, 0, stream>>>(opool, eidx, ew, t1);
    hipMemsetAsync(t2, 0, fsz, stream);
    lgconv<<<(EE + 7) / 8, 256, 0, stream>>>(t1, eidx, ew, t2);
    project<<<(NN + PNODES - 1) / PNODES, 128, 0, stream>>>(
        t2, opool, WtPo, bo, outp + (size_t)NN * HH, 1);
}
